// PicanetG_36876589204042
// MI455X (gfx1250) — compile-verified, matrix operand order checked by reading, not run on hardware
//
#include <hip/hip_runtime.h>
#include <hip/hip_bf16.h>

// ---------------------------------------------------------------------------
// MI455X (gfx1250) implementation: Renet BiLSTM x2 + PicanetG.
// All matmuls on v_wmma_f32_16x16x32_bf16 (bf16 A/B, f32 accum).
// GEMM tiles staged with global_load_async_to_lds_b128 (ASYNCcnt) into
// double-buffered LDS so the HBM stream overlaps the WMMA pipe.
// Workspace layout (bytes):
//   XV  [32768,512] bf16   @ 0      (x transposed to (b,h,w) x c)
//   XH  [32768,512] bf16   @ 32MB   (vertical BiLSTM out, (b,w,h) x 512)
//   HF  [32768,512] bf16   @ 64MB   (horizontal BiLSTM out, (b,w,h) x 512)
//   GF  [32768,1024] f32   @ 96MB   (gate preacts, fwd dir, reused per layer)
//   GB  [32768,1024] f32   @ 224MB  (gate preacts, bwd dir)
//   KQ  [32768,512] f32    @ 352MB  (conv logits, cols>=484 pad)
//   KB  [32768,512] bf16   @ 416MB  (softmaxed kernels)
//   PT  [8,512,512] bf16   @ 448MB  (dilated patch, K padded 484->512)
//   weights (bf16)         @ 452MB
// ---------------------------------------------------------------------------

typedef __attribute__((ext_vector_type(16))) __bf16        v16bf;
typedef __attribute__((ext_vector_type(8)))  float         v8f;
typedef __attribute__((ext_vector_type(4)))  unsigned int  u32x4;

union Frag16 { v16bf v; u32x4 q[2]; };

__device__ __forceinline__ unsigned short f2bf(float f) {
  union { float f; unsigned int u; } cv; cv.f = f;
  unsigned int u = cv.u;
  return (unsigned short)((u + 0x7FFFu + ((u >> 16) & 1u)) >> 16);
}

__device__ __forceinline__ v8f wmma_bf16(const Frag16& a, const Frag16& b, v8f c) {
  return __builtin_amdgcn_wmma_f32_16x16x32_bf16(
      /*neg_a=*/false, a.v, /*neg_b=*/false, b.v,
      /*c_mod=*/(short)0, c, /*reuse_a=*/false, /*reuse_b=*/false);
}

__device__ __forceinline__ float sigm(float x)      { return 1.0f / (1.0f + __expf(-x)); }
__device__ __forceinline__ float tanh_fast(float x) { float e = __expf(2.0f * x); return (e - 1.0f) / (e + 1.0f); }

// Async copy engine: per-lane 16B global -> LDS, tracked by ASYNCcnt.
// LDS operand = low 32 bits of the generic shared address (ISA 10.2).
__device__ __forceinline__ void async_ld_b128(void* lds_dst, const void* gsrc) {
  const unsigned int lds = (unsigned int)(unsigned long long)lds_dst;
  asm volatile("global_load_async_to_lds_b128 %0, %1, off"
               :: "v"(lds), "v"(gsrc) : "memory");
}
__device__ __forceinline__ void wait_asynccnt0() {
  asm volatile("s_wait_asynccnt 0" ::: "memory");
}

// ---------------------------------------------------------------------------
// Generic bf16 GEMM:  C[M,N](f32) = A[M,K](bf16) * Bt[n][k](bf16)^T + bias
// Bt rows are output columns (contiguous K) -> direct WMMA B-fragment loads.
// bswap: B global row = batch*bRowStrideB + ((n&63)<<6 | n>>6)  (PicanetG
// unfold/transpose folded into the fetch).  128x128x32 tile, double-buffered
// LDS filled by async-to-LDS, 8 waves as 4(M) x 2(N), 2x4 accum tiles/wave.
// ---------------------------------------------------------------------------
#define GBM 128
#define GBN 128
#define GBK 32

__global__ __launch_bounds__(256) void gemm_bf16_kernel(
    const unsigned short* __restrict__ A,  int lda, long long aStrideB,
    const unsigned short* __restrict__ Bt, int ldb, int bRowStrideB, int bswap,
    const float* __restrict__ bias, int biasN,
    float* __restrict__ C, int ldc, long long cStrideB, int K)
{
  __shared__ __align__(16) unsigned short As[2][GBM][GBK];
  __shared__ __align__(16) unsigned short Bs[2][GBN][GBK];

  const int tid  = threadIdx.x;
  const int lane = tid & 31;
  const int wid  = tid >> 5;
  const int wm   = (wid & 3) * 32;   // wave M offset (4 waves along M)
  const int wn   = (wid >> 2) * 64;  // wave N offset (2 waves along N)
  const int bm   = blockIdx.y * GBM;
  const int bn   = blockIdx.x * GBN;
  const int bat  = blockIdx.z;

  A += (long long)bat * aStrideB;
  C += (long long)bat * cStrideB;
  const int bRowOff = bat * bRowStrideB;

  v8f acc[2][4];
  {
    v8f z = {0.f,0.f,0.f,0.f,0.f,0.f,0.f,0.f};
    #pragma unroll
    for (int mt = 0; mt < 2; ++mt)
      #pragma unroll
      for (int nt = 0; nt < 4; ++nt) acc[mt][nt] = z;
  }

  const int lr = tid >> 1;          // tile row 0..127 staged by this thread
  const int lc = (tid & 1) * 16;    // 32-byte half of the 64B row
  const int nrow = bn + lr;
  const int grow = (bswap ? (((nrow & 63) << 6) | (nrow >> 6)) : nrow) + bRowOff;
  const unsigned short* agp = A  + (long long)(bm + lr) * lda + lc;
  const unsigned short* bgp = Bt + (long long)grow      * ldb + lc;

  const int kc  = (lane >> 4) * 8;  // per-lane K chunk base (A/B frag layout)
  const int llo = lane & 15;

  // Prologue: async-stage slab 0 into buffer 0
  async_ld_b128(&As[0][lr][lc],     agp);
  async_ld_b128(&As[0][lr][lc + 8], agp + 8);
  async_ld_b128(&Bs[0][lr][lc],     bgp);
  async_ld_b128(&Bs[0][lr][lc + 8], bgp + 8);
  wait_asynccnt0();
  __syncthreads();

  int buf = 0;
  for (int k0 = 0; k0 < K; k0 += GBK) {
    const unsigned short* agn = agp + GBK;
    const unsigned short* bgn = bgp + GBK;
    if (k0 + GBK < K) {  // uniform: prefetch slab k+1 into the other buffer
      async_ld_b128(&As[buf ^ 1][lr][lc],     agn);
      async_ld_b128(&As[buf ^ 1][lr][lc + 8], agn + 8);
      async_ld_b128(&Bs[buf ^ 1][lr][lc],     bgn);
      async_ld_b128(&Bs[buf ^ 1][lr][lc + 8], bgn + 8);
      __builtin_prefetch(agn + GBK, 0, 3);   // global_prefetch_b8, slab k+2
      __builtin_prefetch(bgn + GBK, 0, 3);
    }
    agp = agn; bgp = bgn;

    Frag16 af[2], bfr[4];
    #pragma unroll
    for (int mt = 0; mt < 2; ++mt) {
      af[mt].q[0] = *(const u32x4*)&As[buf][wm + mt * 16 + llo][kc];
      af[mt].q[1] = *(const u32x4*)&As[buf][wm + mt * 16 + llo][kc + 16];
    }
    #pragma unroll
    for (int nt = 0; nt < 4; ++nt) {
      bfr[nt].q[0] = *(const u32x4*)&Bs[buf][wn + nt * 16 + llo][kc];
      bfr[nt].q[1] = *(const u32x4*)&Bs[buf][wn + nt * 16 + llo][kc + 16];
    }
    #pragma unroll
    for (int mt = 0; mt < 2; ++mt)
      #pragma unroll
      for (int nt = 0; nt < 4; ++nt)
        acc[mt][nt] = wmma_bf16(af[mt], bfr[nt], acc[mt][nt]);

    wait_asynccnt0();   // drain this wave's slab-(k+1) copies
    __syncthreads();    // all waves done reading buf and done filling buf^1
    buf ^= 1;
  }

  // Epilogue: C layout lane(0-15)->N=lane, vgpr v->M (+8 for lanes 16-31)
  const int mb = bm + wm + ((lane >> 4) << 3);
  const int nb = bn + wn + llo;
  #pragma unroll
  for (int mt = 0; mt < 2; ++mt)
    #pragma unroll
    for (int nt = 0; nt < 4; ++nt) {
      const int n = nb + nt * 16;
      const float bv = (bias != nullptr && n < biasN) ? bias[n] : 0.0f;
      #pragma unroll
      for (int v = 0; v < 8; ++v)
        C[(long long)(mb + mt * 16 + v) * ldc + n] = acc[mt][nt][v] + bv;
    }
}

// ---------------------------------------------------------------------------
// BiLSTM recurrence. One block = 32 sequences, both directions via blockIdx.y.
// Wave wv owns gate-column slice j in [32*wv, 32*wv+32) for ALL 4 gates
// (8 WMMA N-tiles: i,f,g,o land in the same lane/element) -> cell update is
// pure per-lane VALU; c-state lives in VGPRs; h tile (32x256 bf16) in LDS
// feeds next step's A-fragments via ds_read_b128.
// Gin rows = seq*64 + t (gate preacts, already include x@Wih + bias).
// outMode 1: vertical sweep  -> row = (b*64 + t)*64 + r  (n = b*64+r, t = w)
// outMode 0: horizontal sweep-> row = n*64 + t
// ---------------------------------------------------------------------------
__global__ __launch_bounds__(256) void bilstm_seq_kernel(
    const float* __restrict__ Gf, const float* __restrict__ Gb,
    const unsigned short* __restrict__ Wf, const unsigned short* __restrict__ Wb,
    unsigned short* __restrict__ Out, int outMode)
{
  __shared__ __align__(16) unsigned short hbuf[32][256];  // 16 KB

  const int tid  = threadIdx.x;
  const int lane = tid & 31;
  const int wv   = tid >> 5;
  const int dir  = blockIdx.y;
  const float* __restrict__ Gin          = dir ? Gb : Gf;
  const unsigned short* __restrict__ Whh = dir ? Wb : Wf;
  const int colOff = dir * 256;
  const int n0     = blockIdx.x * 32;
  const int lhi    = lane >> 4;
  const int llo    = lane & 15;

  {
    u32x4 zz = {0u,0u,0u,0u};
    for (int i = tid; i < 1024; i += 256) ((u32x4*)hbuf)[i] = zz;
  }
  __syncthreads();

  float cst[2][2][8];
  #pragma unroll
  for (int mt = 0; mt < 2; ++mt)
    #pragma unroll
    for (int hf = 0; hf < 2; ++hf)
      #pragma unroll
      for (int v = 0; v < 8; ++v) cst[mt][hf][v] = 0.0f;

  for (int tt = 0; tt < 64; ++tt) {
    const int t = dir ? (63 - tt) : tt;

    // Seed accumulators with gate pre-activations (x@Wih + b)
    v8f acc[2][8];
    #pragma unroll
    for (int mt = 0; mt < 2; ++mt)
      #pragma unroll
      for (int g = 0; g < 4; ++g)
        #pragma unroll
        for (int hf = 0; hf < 2; ++hf) {
          const int col = g * 256 + wv * 32 + hf * 16 + llo;
          #pragma unroll
          for (int v = 0; v < 8; ++v) {
            const int m = mt * 16 + v + (lhi << 3);
            acc[mt][g * 2 + hf][v] =
                Gin[((long long)(n0 + m) * 64 + t) * 1024 + col];
          }
        }

    // acc += h @ Whh^T   (K = 256, 8 WMMA k-steps)
    #pragma unroll
    for (int ks = 0; ks < 8; ++ks) {
      const int kc = ks * 32 + (lhi << 3);
      Frag16 af[2];
      #pragma unroll
      for (int mt = 0; mt < 2; ++mt) {
        af[mt].q[0] = *(const u32x4*)&hbuf[mt * 16 + llo][kc];
        af[mt].q[1] = *(const u32x4*)&hbuf[mt * 16 + llo][kc + 16];
      }
      #pragma unroll
      for (int g = 0; g < 4; ++g)
        #pragma unroll
        for (int hf = 0; hf < 2; ++hf) {
          const int ncol = g * 256 + wv * 32 + hf * 16 + llo;
          const unsigned short* bp = Whh + (long long)ncol * 256 + kc;
          Frag16 bfr;
          bfr.q[0] = *(const u32x4*)bp;
          bfr.q[1] = *(const u32x4*)(bp + 16);
          acc[0][g * 2 + hf] = wmma_bf16(af[0], bfr, acc[0][g * 2 + hf]);
          acc[1][g * 2 + hf] = wmma_bf16(af[1], bfr, acc[1][g * 2 + hf]);
        }
    }
    __syncthreads();   // all waves finished reading old h

    // Cell update (per-lane: i,f,g,o share lane/element) + h write-out
    #pragma unroll
    for (int mt = 0; mt < 2; ++mt)
      #pragma unroll
      for (int hf = 0; hf < 2; ++hf)
        #pragma unroll
        for (int v = 0; v < 8; ++v) {
          const float ig = sigm(acc[mt][0 + hf][v]);
          const float fg = sigm(acc[mt][2 + hf][v]);
          const float gg = tanh_fast(acc[mt][4 + hf][v]);
          const float og = sigm(acc[mt][6 + hf][v]);
          const float c  = fg * cst[mt][hf][v] + ig * gg;
          cst[mt][hf][v] = c;
          const float h  = og * tanh_fast(c);

          const int m = mt * 16 + v + (lhi << 3);
          const int j = wv * 32 + hf * 16 + llo;
          const unsigned short hb = f2bf(h);
          hbuf[m][j] = hb;

          const int n = n0 + m;
          const long long row = outMode
              ? ((long long)((n >> 6) * 64 + t) * 64 + (n & 63))
              : ((long long)n * 64 + t);
          Out[row * 512 + colOff + j] = hb;
        }
    __syncthreads();
  }
}

// ---------------------------------------------------------------------------
// x [8,512,64,64] -> XV bf16 [(b,h,w), c] via LDS tile transpose
// ---------------------------------------------------------------------------
__global__ __launch_bounds__(256) void pack_xv_kernel(
    const float* __restrict__ x, unsigned short* __restrict__ XV)
{
  __shared__ float tile[64][65];
  const int c0 = blockIdx.x * 64, h = blockIdx.y, b = blockIdx.z;
  for (int i = threadIdx.x; i < 4096; i += 256) {
    const int cl = i >> 6, wl = i & 63;
    tile[cl][wl] = x[(((long long)b * 512 + c0 + cl) * 64 + h) * 64 + wl];
  }
  __syncthreads();
  for (int i = threadIdx.x; i < 4096; i += 256) {
    const int wl = i >> 6, cl = i & 63;
    XV[((long long)(b * 64 + h) * 64 + wl) * 512 + c0 + cl] = f2bf(tile[cl][wl]);
  }
}

// x[:, :, ::3, ::3] -> PT bf16 [8,512,512] (K padded 484->512 with zeros)
__global__ __launch_bounds__(512) void pack_patch_kernel(
    const float* __restrict__ x, unsigned short* __restrict__ PT)
{
  const int b = blockIdx.x >> 9, c = blockIdx.x & 511, k = threadIdx.x;
  float v = 0.0f;
  if (k < 484) {
    const int r = k / 22, cc = k % 22;
    v = x[(((long long)b * 512 + c) * 64 + 3 * r) * 64 + 3 * cc];
  }
  PT[((long long)b * 512 + c) * 512 + k] = f2bf(v);
}

// Row-wise softmax over 484 logits, one wave32 per row, out bf16 (pad zeroed)
__global__ __launch_bounds__(256) void softmax_kernel(
    const float* __restrict__ KQ, unsigned short* __restrict__ KB)
{
  const int row  = blockIdx.x * 8 + (threadIdx.x >> 5);
  const int lane = threadIdx.x & 31;
  const float* src = KQ + (long long)row * 512;
  float vals[16];
  int cnt = 0;
  float mx = -1e30f;
  for (int c = lane; c < 484; c += 32) { float v = src[c]; vals[cnt++] = v; mx = fmaxf(mx, v); }
  #pragma unroll
  for (int o = 16; o > 0; o >>= 1) mx = fmaxf(mx, __shfl_xor(mx, o, 32));
  float s = 0.0f;
  for (int i = 0; i < cnt; ++i) { vals[i] = __expf(vals[i] - mx); s += vals[i]; }
  #pragma unroll
  for (int o = 16; o > 0; o >>= 1) s += __shfl_xor(s, o, 32);
  const float inv = 1.0f / s;
  unsigned short* dst = KB + (long long)row * 512;
  cnt = 0;
  for (int c = lane; c < 484; c += 32) dst[c] = f2bf(vals[cnt++] * inv);
  for (int c = 484 + lane; c < 512; c += 32) dst[c] = 0;
}

// f32 -> bf16 weight converts
__global__ __launch_bounds__(256) void cvt_bf16_kernel(
    const float* __restrict__ s, unsigned short* __restrict__ d, int n)
{
  const int i = blockIdx.x * 256 + threadIdx.x;
  if (i < n) d[i] = f2bf(s[i]);
}
__global__ __launch_bounds__(256) void cvt_convw_kernel(
    const float* __restrict__ s, unsigned short* __restrict__ d)
{
  const int i = blockIdx.x * 256 + threadIdx.x;   // over 512*512
  const int o = i >> 9, c = i & 511;
  d[i] = (o < 484) ? f2bf(s[o * 512 + c]) : (unsigned short)0;
}

// ---------------------------------------------------------------------------
extern "C" void kernel_launch(void* const* d_in, const int* in_sizes, int n_in,
                              void* d_out, int out_size, void* d_ws, size_t ws_size,
                              hipStream_t stream)
{
  const float* x       = (const float*)d_in[0];
  const float* v_wih_f = (const float*)d_in[1];
  const float* v_whh_f = (const float*)d_in[2];
  const float* v_b_f   = (const float*)d_in[3];
  const float* v_wih_b = (const float*)d_in[4];
  const float* v_whh_b = (const float*)d_in[5];
  const float* v_b_b   = (const float*)d_in[6];
  const float* h_wih_f = (const float*)d_in[7];
  const float* h_whh_f = (const float*)d_in[8];
  const float* h_b_f   = (const float*)d_in[9];
  const float* h_wih_b = (const float*)d_in[10];
  const float* h_whh_b = (const float*)d_in[11];
  const float* h_b_b   = (const float*)d_in[12];
  const float* conv_w  = (const float*)d_in[13];
  const float* conv_b  = (const float*)d_in[14];
  float* out = (float*)d_out;

  char* ws = (char*)d_ws;
  const size_t MB = 1024ull * 1024ull;
  unsigned short* XV = (unsigned short*)(ws + 0);
  unsigned short* XH = (unsigned short*)(ws + 32 * MB);
  unsigned short* HF = (unsigned short*)(ws + 64 * MB);
  float*          GF = (float*)(ws + 96 * MB);
  float*          GB = (float*)(ws + 224 * MB);
  float*          KQ = (float*)(ws + 352 * MB);
  unsigned short* KB = (unsigned short*)(ws + 416 * MB);
  unsigned short* PT = (unsigned short*)(ws + 448 * MB);
  size_t wo = 452 * MB;
  unsigned short* W_vwf = (unsigned short*)(ws + wo); wo += 1024 * 512 * 2;
  unsigned short* W_vwb = (unsigned short*)(ws + wo); wo += 1024 * 512 * 2;
  unsigned short* W_vuf = (unsigned short*)(ws + wo); wo += 1024 * 256 * 2;
  unsigned short* W_vub = (unsigned short*)(ws + wo); wo += 1024 * 256 * 2;
  unsigned short* W_hwf = (unsigned short*)(ws + wo); wo += 1024 * 512 * 2;
  unsigned short* W_hwb = (unsigned short*)(ws + wo); wo += 1024 * 512 * 2;
  unsigned short* W_huf = (unsigned short*)(ws + wo); wo += 1024 * 256 * 2;
  unsigned short* W_hub = (unsigned short*)(ws + wo); wo += 1024 * 256 * 2;
  unsigned short* W_cw  = (unsigned short*)(ws + wo); wo += 512 * 512 * 2;

  // Weight converts (f32 -> bf16; conv_w padded to 512 output rows)
  cvt_bf16_kernel<<<2048, 256, 0, stream>>>(v_wih_f, W_vwf, 1024 * 512);
  cvt_bf16_kernel<<<2048, 256, 0, stream>>>(v_wih_b, W_vwb, 1024 * 512);
  cvt_bf16_kernel<<<1024, 256, 0, stream>>>(v_whh_f, W_vuf, 1024 * 256);
  cvt_bf16_kernel<<<1024, 256, 0, stream>>>(v_whh_b, W_vub, 1024 * 256);
  cvt_bf16_kernel<<<2048, 256, 0, stream>>>(h_wih_f, W_hwf, 1024 * 512);
  cvt_bf16_kernel<<<2048, 256, 0, stream>>>(h_wih_b, W_hwb, 1024 * 512);
  cvt_bf16_kernel<<<1024, 256, 0, stream>>>(h_whh_f, W_huf, 1024 * 256);
  cvt_bf16_kernel<<<1024, 256, 0, stream>>>(h_whh_b, W_hub, 1024 * 256);
  cvt_convw_kernel<<<1024, 256, 0, stream>>>(conv_w, W_cw);

  // x -> XV [(b,h,w), c] bf16
  pack_xv_kernel<<<dim3(8, 64, 8), 256, 0, stream>>>(x, XV);

  // Vertical sweep: gate preacts (time-parallel) then recurrence
  gemm_bf16_kernel<<<dim3(8, 256, 1), 256, 0, stream>>>(
      XV, 512, 0, W_vwf, 512, 0, 0, v_b_f, 1024, GF, 1024, 0, 512);
  gemm_bf16_kernel<<<dim3(8, 256, 1), 256, 0, stream>>>(
      XV, 512, 0, W_vwb, 512, 0, 0, v_b_b, 1024, GB, 1024, 0, 512);
  bilstm_seq_kernel<<<dim3(16, 2), 256, 0, stream>>>(GF, GB, W_vuf, W_vub, XH, 1);

  // Horizontal sweep
  gemm_bf16_kernel<<<dim3(8, 256, 1), 256, 0, stream>>>(
      XH, 512, 0, W_hwf, 512, 0, 0, h_b_f, 1024, GF, 1024, 0, 512);
  gemm_bf16_kernel<<<dim3(8, 256, 1), 256, 0, stream>>>(
      XH, 512, 0, W_hwb, 512, 0, 0, h_b_b, 1024, GB, 1024, 0, 512);
  bilstm_seq_kernel<<<dim3(16, 2), 256, 0, stream>>>(GF, GB, W_huf, W_hub, HF, 0);

  // 1x1 conv -> logits (N padded to 512), softmax, patch pack
  gemm_bf16_kernel<<<dim3(4, 256, 1), 256, 0, stream>>>(
      HF, 512, 0, W_cw, 512, 0, 0, conv_b, 484, KQ, 512, 0, 512);
  softmax_kernel<<<4096, 256, 0, stream>>>(KQ, KB);
  pack_patch_kernel<<<4096, 512, 0, stream>>>(x, PT);

  // Global attention: out[b] = patch[b] @ kern[b]; unfold-transpose folded
  // into B row map (bswap), K padded 484->512 with zero A columns.
  gemm_bf16_kernel<<<dim3(32, 4, 8), 256, 0, stream>>>(
      PT, 512, (long long)512 * 512,
      KB, 512, 4096, 1,
      nullptr, 0,
      out, 4096, (long long)512 * 4096, 512);
}
